// MGNN_36240934044227
// MI455X (gfx1250) — matrix operand order repaired and checked by reading, not compiled, not run on hardware
//
#include <hip/hip_runtime.h>
#include <math.h>

typedef unsigned short u16;
typedef unsigned int   u32;
typedef __attribute__((ext_vector_type(16))) __bf16 v16bf;
typedef __attribute__((ext_vector_type(8)))  float  v8f;
typedef __attribute__((ext_vector_type(4)))  int    v4i;

#define NNODE 16000
#define NEDGE 512000
#define NFEATC 200
#define NHID  256
#define K0C   10
#define K1C   5
#define CATW  768   // 200 + 256 + 256 = 712, padded to 768 (multiple of 64)
#define REPW  768   // 3*256

// CDNA5 async global->LDS copy path (probe via __has_builtin; fallback compiles anywhere)
#if defined(__gfx1250__) && __has_builtin(__builtin_amdgcn_global_load_async_to_lds_b128) && \
    __has_builtin(__builtin_amdgcn_s_wait_asynccnt)
#define USE_ASYNC 1
#else
#define USE_ASYNC 0
#endif

// ---------------- helpers ----------------

__device__ __forceinline__ u16 f2b(float f) {           // f32 -> bf16 bits, RNE
  union { float f; u32 u; } x; x.f = f;
  u32 r = x.u + 0x7fffu + ((x.u >> 16) & 1u);
  return (u16)(r >> 16);
}

__device__ __forceinline__ float gumbel_noise(u32 a, u32 b) {
  u32 s = a * 0x9E3779B9u ^ (b + 0x85EBCA6Bu);
  s ^= s >> 16; s *= 0x7FEB352Du; s ^= s >> 15; s *= 0x846CA68Bu; s ^= s >> 16;
  float u = (float)(s >> 8) * (1.0f / 16777216.0f) + 1e-10f;
  return -logf(-logf(u));
}

// ---------------- f32 -> bf16 convert with zero padding ----------------
__global__ void k_convert_pad(const float* __restrict__ src, int srcRows, int srcCols,
                              u16* __restrict__ dst, int dstRows, int dstCols, int dstStride) {
  long i = (long)blockIdx.x * blockDim.x + threadIdx.x;
  long total = (long)dstRows * dstCols;
  if (i >= total) return;
  int r = (int)(i / dstCols), c = (int)(i % dstCols);
  float v = (r < srcRows && c < srcCols) ? src[(long)r * srcCols + c] : 0.0f;
  dst[(long)r * dstStride + c] = f2b(v);
}

// ---------------- WMMA bf16 GEMM: C = act(A[MxK] @ W[KxN] + bias) ----------------
// Block: 256 threads = 8 waves; tile 128(M) x 16*NT(N); K step 64 via LDS.
// A tile staged with gfx1250 async global->LDS (ASYNCcnt) when available.
// Requires: M%128==0, N%(16*NT)==0, K%64==0. A,W bf16(u16); out f32 and/or bf16.
#define BMT 128
#define BKT 64
#define APAD 72   // LDS stride (bf16 elems): 72*2=144B, 16B-aligned rows
template <int NT>
__global__ __launch_bounds__(256)
void k_gemm(const u16* __restrict__ A, int lda,
            const u16* __restrict__ W, int ldw,
            const float* __restrict__ bias,
            float* __restrict__ Cf, int ldcf,
            u16* __restrict__ Cb, int ldcb,
            int M, int N, int K, int act) {
  constexpr int BN = 16 * NT;
  constexpr int NB = BN / 8;            // B pair-copy iterations per thread
  __shared__ u16 sA[BMT * APAD];        // 18432 B
  __shared__ u16 sB[BN * APAD];         // transposed: [n][k]; NT=4 -> 9216 B
  const int tid  = threadIdx.x;
  const int wave = tid >> 5;
  const int lane = tid & 31;
  const int m0 = blockIdx.x * BMT;
  const int n0 = blockIdx.y * BN;

  v8f acc[NT];
#pragma unroll
  for (int t = 0; t < NT; ++t) acc[t] = v8f{0.f, 0.f, 0.f, 0.f, 0.f, 0.f, 0.f, 0.f};

  // Precompute per-thread copy slots (advance by constant strides each K step).
  const u16* agp[4]; u16* alp[4];
#pragma unroll
  for (int it = 0; it < 4; ++it) {
    int chunk = tid + it * 256;          // 0..1023 : A tile = 1024 x 16B chunks
    int r = chunk >> 3;                  // 8 chunks per 64-elem row
    int c = (chunk & 7) * 8;
    agp[it] = A + (size_t)(m0 + r) * lda + c;
    alp[it] = &sA[r * APAD + c];
  }
  const u16* bgp[NB]; u16* blp[NB];
#pragma unroll
  for (int it = 0; it < NB; ++it) {
    int idx = tid + it * 256;            // over 32*BN pair entries
    int kk = idx / BN;                   // 0..31 (k pair index)
    int n  = idx % BN;
    bgp[it] = W + (size_t)(2 * kk) * ldw + n0 + n;
    blp[it] = &sB[n * APAD + 2 * kk];
  }
  const size_t bstep = (size_t)BKT * ldw;

  for (int k0 = 0; k0 < K; k0 += BKT) {
#if USE_ASYNC
    // A tile: async DMA global->LDS, no VGPR round trip, tracked by ASYNCcnt
#pragma unroll
    for (int it = 0; it < 4; ++it)
      __builtin_amdgcn_global_load_async_to_lds_b128(
          (__attribute__((address_space(1))) v4i*)agp[it],
          (__attribute__((address_space(3))) v4i*)alp[it], 0, 0);
#else
    uint4 va[4];
#pragma unroll
    for (int it = 0; it < 4; ++it) va[it] = *(const uint4*)agp[it];   // all loads in flight
#endif
    // B tile: load all half-words first (pipelined), then pack k-pairs -> b32 LDS stores
    u32 blo[NB], bhi[NB];
#pragma unroll
    for (int it = 0; it < NB; ++it) { blo[it] = bgp[it][0]; bhi[it] = bgp[it][ldw]; }
#pragma unroll
    for (int it = 0; it < NB; ++it) *(u32*)blp[it] = blo[it] | (bhi[it] << 16);
#if !USE_ASYNC
#pragma unroll
    for (int it = 0; it < 4; ++it) *(uint4*)alp[it] = va[it];
    if (k0 + BKT < K) __builtin_prefetch(agp[0] + BKT, 0, 3);   // gfx1250 global_prefetch
#endif
#pragma unroll
    for (int it = 0; it < 4; ++it) agp[it] += BKT;
#pragma unroll
    for (int it = 0; it < NB; ++it) bgp[it] += bstep;
#if USE_ASYNC
    __builtin_amdgcn_s_wait_asynccnt(0);
#endif
    __syncthreads();
#pragma unroll
    for (int ks = 0; ks < BKT; ks += 32) {
      // A fragment: lane<16 -> row=lane, K {0..7,16..23}; lane>=16 -> row=lane-16, K {8..15,24..31}
      int arow = wave * 16 + (lane & 15);
      int ak   = ks + ((lane < 16) ? 0 : 8);
      union { uint4 u[2]; v16bf v; } af;
      af.u[0] = *(const uint4*)(&sA[arow * APAD + ak]);
      af.u[1] = *(const uint4*)(&sA[arow * APAD + ak + 16]);
      int bk = ks + ((lane < 16) ? 0 : 8);
#pragma unroll
      for (int t = 0; t < NT; ++t) {
        // B fragment: lane<16 -> col, K {0..7,16..23}; lane>=16 -> col, K {8..15,24..31}
        int bn = (lane & 15) + t * 16;
        union { uint4 u[2]; v16bf v; } bfv;
        bfv.u[0] = *(const uint4*)(&sB[bn * APAD + bk]);
        bfv.u[1] = *(const uint4*)(&sB[bn * APAD + bk + 16]);
        acc[t] = __builtin_amdgcn_wmma_f32_16x16x32_bf16(false, af.v, false, bfv.v,
                                                         (short)0, acc[t], false, false);
      }
    }
    __syncthreads();
  }
  // C/D layout: VGPR r: lanes 0-15 -> M=r, N=lane; lanes 16-31 -> M=8+r, N=lane-16
  const int rowbase = m0 + wave * 16 + ((lane < 16) ? 0 : 8);
  const int colbase = n0 + (lane & 15);
  float vout[NT][8];
#pragma unroll
  for (int t = 0; t < NT; ++t) {
    float bv = bias ? bias[colbase + 16 * t] : 0.0f;
#pragma unroll
    for (int r = 0; r < 8; ++r) {
      float v = acc[t][r] + bv;
      vout[t][r] = (act == 1) ? fmaxf(v, 0.0f) : v;
    }
  }
  if (Cf) {
    float* base = Cf + (size_t)rowbase * ldcf + colbase;
#pragma unroll
    for (int t = 0; t < NT; ++t)
#pragma unroll
      for (int r = 0; r < 8; ++r) base[(size_t)r * ldcf + 16 * t] = vout[t][r];
  }
  if (Cb) {
    u16* base = Cb + (size_t)rowbase * ldcb + colbase;
#pragma unroll
    for (int t = 0; t < NT; ++t)
#pragma unroll
      for (int r = 0; r < 8; ++r) base[(size_t)r * ldcb + 16 * t] = f2b(vout[t][r]);
  }
}

// ---------------- GCN normalization / propagation ----------------
__global__ void k_deg_init(float* deg) {
  int i = blockIdx.x * blockDim.x + threadIdx.x;
  if (i < NNODE) deg[i] = 1.0f;                   // self-loop weight
}
__global__ void k_deg_edges(const int* __restrict__ dst, const float* __restrict__ ew, float* deg) {
  int e = blockIdx.x * blockDim.x + threadIdx.x;
  if (e < NEDGE) atomicAdd(&deg[dst[e]], ew[e]);
}
__global__ void k_dinv(float* deg) {
  int i = blockIdx.x * blockDim.x + threadIdx.x;
  if (i < NNODE) { float d = deg[i]; deg[i] = (d > 0.f) ? rsqrtf(d) : 0.f; }
}
// agg[i,f] = dinv[i]^2 * XW[i,f] + bias[f]   (self loop, ew=1)
__global__ void k_selfloop(const float* __restrict__ XW, const float* __restrict__ dinv,
                           const float* __restrict__ bias, float* __restrict__ agg) {
  long i = (long)blockIdx.x * blockDim.x + threadIdx.x;
  if (i >= (long)NNODE * NHID) return;
  int row = (int)(i >> 8), f = (int)(i & 255);
  float di = dinv[row];
  agg[i] = di * di * XW[i] + bias[f];
}
// one wave per edge; lane handles 8 features; agg fits in L2 so atomics stay on-chip
__global__ void k_edge_prop(const int* __restrict__ src, const int* __restrict__ dst,
                            const float* __restrict__ ew, const float* __restrict__ dinv,
                            const float* __restrict__ h, float* __restrict__ agg) {
  long t = (long)blockIdx.x * blockDim.x + threadIdx.x;
  long e = t >> 5;
  if (e >= NEDGE) return;
  int lane = (int)(t & 31);
  int s = src[e], d = dst[e];
  float coef = dinv[s] * ew[e] * dinv[d];
  const float4* hs = (const float4*)(h + (size_t)s * NHID + lane * 8);
  float4 a = hs[0], b = hs[1];
  float* ad = agg + (size_t)d * NHID + lane * 8;
  atomicAdd(ad + 0, coef * a.x); atomicAdd(ad + 1, coef * a.y);
  atomicAdd(ad + 2, coef * a.z); atomicAdd(ad + 3, coef * a.w);
  atomicAdd(ad + 4, coef * b.x); atomicAdd(ad + 5, coef * b.y);
  atomicAdd(ad + 6, coef * b.z); atomicAdd(ad + 7, coef * b.w);
}

// LayerNorm + ReLU over 256-wide rows, emit bf16 into strided slice
__global__ __launch_bounds__(256)
void k_ln_relu(const float* __restrict__ agg, const float* __restrict__ g,
               const float* __restrict__ b, u16* __restrict__ out, int stride) {
  __shared__ float red[256];
  int row = blockIdx.x, t = threadIdx.x;
  float v = agg[(size_t)row * NHID + t];
  red[t] = v; __syncthreads();
  for (int s = 128; s > 0; s >>= 1) { if (t < s) red[t] += red[t + s]; __syncthreads(); }
  float mean = red[0] * (1.0f / NHID); __syncthreads();
  float d = v - mean;
  red[t] = d * d; __syncthreads();
  for (int s = 128; s > 0; s >>= 1) { if (t < s) red[t] += red[t + s]; __syncthreads(); }
  float var = red[0] * (1.0f / NHID);
  float o = d * rsqrtf(var + 1e-5f) * g[t] + b[t];
  out[(size_t)row * stride + t] = f2b(fmaxf(o, 0.0f));
}

// ---------------- pooling (assign is hard one-hot => index ops) ----------------
__global__ void k_gax0(const float* __restrict__ logits, const float* __restrict__ bias,
                       int* __restrict__ c, int n) {
  int i = blockIdx.x * blockDim.x + threadIdx.x;
  if (i >= n) return;
  float best = -1e30f; int bj = 0;
  for (int j = 0; j < K0C; ++j) {
    float v = logits[(size_t)i * 16 + j] + bias[j] + gumbel_noise((u32)i * 16u + j, 0x1234567u);
    if (v > best) { best = v; bj = j; }
  }
  c[i] = bj;
}
__global__ void k_zero(float* p, long n) {
  long i = (long)blockIdx.x * blockDim.x + threadIdx.x;
  if (i < n) p[i] = 0.0f;
}
__global__ void k_xc_accum(const float* __restrict__ bottom, const int* __restrict__ c0,
                           float* __restrict__ xc0) {
  long i = (long)blockIdx.x * blockDim.x + threadIdx.x;
  if (i >= (long)NNODE * NHID) return;
  int row = (int)(i >> 8), f = (int)(i & 255);
  atomicAdd(&xc0[c0[row] * NHID + f], bottom[i]);
}
__global__ __launch_bounds__(256) void k_l2norm_rows(float* X) {   // rows of width 256
  __shared__ float red[256];
  int row = blockIdx.x, t = threadIdx.x;
  float v = X[(size_t)row * NHID + t];
  red[t] = v * v; __syncthreads();
  for (int s = 128; s > 0; s >>= 1) { if (t < s) red[t] += red[t + s]; __syncthreads(); }
  float nrm = fmaxf(sqrtf(red[0]), 1e-12f);
  X[(size_t)row * NHID + t] = v / nrm;
}
__global__ void k_adj0e(const int* __restrict__ src, const int* __restrict__ dst,
                        const int* __restrict__ c0, float* __restrict__ adj) {
  int e = blockIdx.x * blockDim.x + threadIdx.x;
  if (e < NEDGE) atomicAdd(&adj[c0[src[e]] * K0C + c0[dst[e]]], 1.0f);
}
__global__ __launch_bounds__(128) void k_scale_sum(float* A, int n) {  // A /= sum(A)
  __shared__ float red[128];
  int t = threadIdx.x;
  float s = 0.f;
  for (int i = t; i < n; i += 128) s += A[i];
  red[t] = s; __syncthreads();
  for (int k = 64; k > 0; k >>= 1) { if (t < k) red[t] += red[t + k]; __syncthreads(); }
  float inv = (red[0] != 0.f) ? 1.0f / red[0] : 0.f;
  for (int i = t; i < n; i += 128) A[i] *= inv;
}
// tiny fp32 matmul: C = act(A[MxK]@B[KxN] + bias); act: 0=none, 2=tanh
__global__ void k_small_mm(const float* __restrict__ A, const float* __restrict__ B,
                           const float* __restrict__ bias, float* __restrict__ C,
                           int M, int Kk, int Nn, int act) {
  int i = blockIdx.x * blockDim.x + threadIdx.x;
  if (i >= M * Nn) return;
  int m = i / Nn, n = i % Nn;
  float s = bias ? bias[n] : 0.0f;
  for (int k = 0; k < Kk; ++k) s += A[m * Kk + k] * B[k * Nn + n];
  if (act == 2) s = tanhf(s);
  C[i] = s;
}
__global__ void k_gax1(const float* __restrict__ logits, int* __restrict__ c) {
  int r = threadIdx.x; if (r >= K0C) return;
  float best = -1e30f; int bj = 0;
  for (int j = 0; j < K1C; ++j) {
    float v = logits[r * K1C + j] + gumbel_noise((u32)r * 8u + j, 0x89ABCDEFu);
    if (v > best) { best = v; bj = j; }
  }
  c[r] = bj;
}
__global__ __launch_bounds__(256) void k_xc1(const float* __restrict__ lat1,
                                             const int* __restrict__ c1, float* __restrict__ xc1) {
  int f = threadIdx.x;
  float a[K1C]; for (int k = 0; k < K1C; ++k) a[k] = 0.f;
  for (int r = 0; r < K0C; ++r) a[c1[r]] += lat1[r * NHID + f];
  for (int k = 0; k < K1C; ++k) xc1[k * NHID + f] = a[k];
}
__global__ void k_adj1(const float* __restrict__ adj0, const int* __restrict__ c1,
                       float* __restrict__ adj1) {
  int t = threadIdx.x; if (t >= K1C * K1C) return;
  int a = t / K1C, b = t % K1C; float s = 0.f;
  for (int i = 0; i < K0C; ++i)
    for (int j = 0; j < K0C; ++j)
      if (c1[i] == a && c1[j] == b) s += adj0[i * K0C + j];
  adj1[t] = s;
}
__global__ void k_rep_lat(const float* __restrict__ lat1, const float* __restrict__ lat2,
                          const int* __restrict__ c0, const int* __restrict__ c1,
                          u16* __restrict__ rep) {
  long i = (long)blockIdx.x * blockDim.x + threadIdx.x;
  if (i >= (long)NNODE * NHID) return;
  int row = (int)(i >> 8), f = (int)(i & 255);
  int k0 = c0[row];
  rep[(size_t)row * REPW + 256 + f] = f2b(lat1[k0 * NHID + f]);
  rep[(size_t)row * REPW + 512 + f] = f2b(lat2[c1[k0] * NHID + f]);
}
__global__ void k_lsm16(const float* __restrict__ in, float* __restrict__ out, int n) {
  int i = blockIdx.x * blockDim.x + threadIdx.x;
  if (i >= n) return;
  float v[16]; float mx = -1e30f;
  for (int j = 0; j < 16; ++j) { v[j] = in[(size_t)i * 16 + j]; mx = fmaxf(mx, v[j]); }
  float s = 0.f;
  for (int j = 0; j < 16; ++j) s += expf(v[j] - mx);
  float l = logf(s);
  for (int j = 0; j < 16; ++j) out[(size_t)i * 16 + j] = v[j] - mx - l;
}

// ---------------- host orchestration ----------------
extern "C" void kernel_launch(void* const* d_in, const int* in_sizes, int n_in,
                              void* d_out, int out_size, void* d_ws, size_t ws_size,
                              hipStream_t stream) {
  const float* x       = (const float*)d_in[0];
  const int*   ei      = (const int*)d_in[1];
  const float* ew      = (const float*)d_in[2];
  const float* conv1_W = (const float*)d_in[3];
  const float* conv1_b = (const float*)d_in[4];
  const float* conv2_W = (const float*)d_in[5];
  const float* conv2_b = (const float*)d_in[6];
  const float* ln1_g   = (const float*)d_in[7];
  const float* ln1_b   = (const float*)d_in[8];
  const float* ln2_g   = (const float*)d_in[9];
  const float* ln2_b   = (const float*)d_in[10];
  const float* efc1_W  = (const float*)d_in[11];
  const float* efc1_b  = (const float*)d_in[12];
  const float* efc2_W  = (const float*)d_in[13];
  const float* efc2_b  = (const float*)d_in[14];
  const float* lin0_W  = (const float*)d_in[15];
  const float* lin0_b  = (const float*)d_in[16];
  const float* lin1_W  = (const float*)d_in[17];
  const float* lin1_b  = (const float*)d_in[18];
  const float* enc0_W  = (const float*)d_in[19];
  const float* enc0_b  = (const float*)d_in[20];
  const float* enc1_W  = (const float*)d_in[21];
  const float* enc1_b  = (const float*)d_in[22];
  const float* fc1_W   = (const float*)d_in[23];
  const float* fc1_b   = (const float*)d_in[24];
  const float* fc2_W   = (const float*)d_in[25];
  const float* fc2_b   = (const float*)d_in[26];
  float* out = (float*)d_out;
  (void)in_sizes; (void)n_in; (void)out_size; (void)ws_size;

  char* ws = (char*)d_ws;
  size_t off = 0;
  auto take = [&](size_t bytes) -> void* {
    size_t o = (off + 255) & ~(size_t)255; off = o + bytes; return (void*)(ws + o);
  };
  u16*   xpad    = (u16*)take((size_t)NNODE * 256 * 2);
  u16*   w1      = (u16*)take(256 * 256 * 2);
  u16*   w2      = (u16*)take(256 * 256 * 2);
  u16*   efc1w   = (u16*)take(768 * 256 * 2);
  u16*   efc2w   = (u16*)take(256 * 256 * 2);
  u16*   lin0w   = (u16*)take(256 * 16 * 2);
  u16*   fc1w    = (u16*)take(768 * 256 * 2);
  u16*   fc2w    = (u16*)take(256 * 16 * 2);
  float* XW      = (float*)take((size_t)NNODE * NHID * 4);
  float* dinv    = (float*)take((size_t)NNODE * 4);
  float* agg     = (float*)take((size_t)NNODE * NHID * 4);
  u16*   cat     = (u16*)take((size_t)NNODE * CATW * 2);
  u16*   midb    = (u16*)take((size_t)NNODE * NHID * 2);
  float* bottom  = (float*)take((size_t)NNODE * NHID * 4);
  u16*   rep     = (u16*)take((size_t)NNODE * REPW * 2);
  float* logits0 = (float*)take((size_t)NNODE * 16 * 4);
  int*   c0      = (int*)take((size_t)NNODE * 4);
  float* xc0     = (float*)take(K0C * NHID * 4);
  float* adj0    = (float*)take(K0C * K0C * 4);
  float* tmpA    = (float*)take(K0C * NHID * 4);
  float* lat1    = (float*)take(K0C * NHID * 4);
  float* logits1 = (float*)take(K0C * K1C * 4);
  int*   c1      = (int*)take(64);
  float* xc1     = (float*)take(K1C * NHID * 4);
  float* adj1    = (float*)take(K1C * K1C * 4);
  float* tmpB    = (float*)take(K1C * NHID * 4);
  float* lat2    = (float*)take(K1C * NHID * 4);
  u16*   mid2    = (u16*)take((size_t)NNODE * NHID * 2);
  float* out16   = (float*)take((size_t)NNODE * 16 * 4);

  const int TB = 256;
  auto g1 = [&](long n) { return dim3((unsigned)((n + TB - 1) / TB)); };
  dim3 gg256(NNODE / BMT, NHID / 64);   // 125 x 4   (NT=4 -> 64 cols/block)
  dim3 gg16(NNODE / BMT, 1);            // 125 x 1   (NT=1)

  // feature / weight conversion to bf16 with zero padding
  k_convert_pad<<<g1((long)NNODE * 256), TB, 0, stream>>>(x, NNODE, NFEATC, xpad, NNODE, 256, 256);
  k_convert_pad<<<g1((long)NNODE * 200), TB, 0, stream>>>(x, NNODE, NFEATC, cat, NNODE, 200, CATW);
  k_convert_pad<<<g1((long)NNODE * 56), TB, 0, stream>>>(x, 0, 0, cat + 712, NNODE, 56, CATW);
  k_convert_pad<<<g1(256 * 256), TB, 0, stream>>>(conv1_W, 200, 256, w1, 256, 256, 256);
  k_convert_pad<<<g1(256 * 256), TB, 0, stream>>>(conv2_W, 256, 256, w2, 256, 256, 256);
  k_convert_pad<<<g1(768 * 256), TB, 0, stream>>>(efc1_W, 712, 256, efc1w, 768, 256, 256);
  k_convert_pad<<<g1(256 * 256), TB, 0, stream>>>(efc2_W, 256, 256, efc2w, 256, 256, 256);
  k_convert_pad<<<g1(256 * 16), TB, 0, stream>>>(lin0_W, 256, 10, lin0w, 256, 16, 16);
  k_convert_pad<<<g1(768 * 256), TB, 0, stream>>>(fc1_W, 768, 256, fc1w, 768, 256, 256);
  k_convert_pad<<<g1(256 * 16), TB, 0, stream>>>(fc2_W, 256, 16, fc2w, 256, 16, 16);

  // symmetric GCN normalization
  k_deg_init<<<g1(NNODE), TB, 0, stream>>>(dinv);
  k_deg_edges<<<g1(NEDGE), TB, 0, stream>>>(ei + NEDGE, ew, dinv);
  k_dinv<<<g1(NNODE), TB, 0, stream>>>(dinv);

  // GCN layer 1
  k_gemm<4><<<gg256, 256, 0, stream>>>(xpad, 256, w1, 256, nullptr, XW, 256, nullptr, 0, NNODE, 256, 256, 0);
  k_selfloop<<<g1((long)NNODE * NHID), TB, 0, stream>>>(XW, dinv, conv1_b, agg);
  k_edge_prop<<<g1((long)NEDGE * 32), TB, 0, stream>>>(ei, ei + NEDGE, ew, dinv, XW, agg);
  k_ln_relu<<<NNODE, 256, 0, stream>>>(agg, ln1_g, ln1_b, cat + 200, CATW);

  // GCN layer 2 (A = h1 slice of cat, lda=CATW)
  k_gemm<4><<<gg256, 256, 0, stream>>>(cat + 200, CATW, w2, 256, nullptr, XW, 256, nullptr, 0, NNODE, 256, 256, 0);
  k_selfloop<<<g1((long)NNODE * NHID), TB, 0, stream>>>(XW, dinv, conv2_b, agg);
  k_edge_prop<<<g1((long)NEDGE * 32), TB, 0, stream>>>(ei, ei + NEDGE, ew, dinv, XW, agg);
  k_ln_relu<<<NNODE, 256, 0, stream>>>(agg, ln2_g, ln2_b, cat + 456, CATW);

  // encoder MLP: mid = relu(cat@efc1+b); bottom = mid@efc2+b (f32 + bf16 into rep[:,0:256])
  k_gemm<4><<<gg256, 256, 0, stream>>>(cat, CATW, efc1w, 256, efc1_b, nullptr, 0, midb, 256, NNODE, 256, 768, 1);
  k_gemm<4><<<gg256, 256, 0, stream>>>(midb, 256, efc2w, 256, efc2_b, bottom, 256, rep, REPW, NNODE, 256, 256, 0);

  // pooling level 0: logits, hard gumbel argmax, cluster reductions
  k_gemm<1><<<gg16, 256, 0, stream>>>(rep, REPW, lin0w, 16, nullptr, logits0, 16, nullptr, 0, NNODE, 16, 256, 0);
  k_gax0<<<g1(NNODE), TB, 0, stream>>>(logits0, lin0_b, c0, NNODE);
  k_zero<<<g1(K0C * NHID), TB, 0, stream>>>(xc0, K0C * NHID);
  k_xc_accum<<<g1((long)NNODE * NHID), TB, 0, stream>>>(bottom, c0, xc0);
  k_l2norm_rows<<<K0C, 256, 0, stream>>>(xc0);
  k_zero<<<1, 128, 0, stream>>>(adj0, K0C * K0C);
  k_adj0e<<<g1(NEDGE), TB, 0, stream>>>(ei, ei + NEDGE, c0, adj0);
  k_scale_sum<<<1, 128, 0, stream>>>(adj0, K0C * K0C);
  k_small_mm<<<g1(K0C * NHID), TB, 0, stream>>>(adj0, xc0, nullptr, tmpA, K0C, K0C, NHID, 0);
  k_small_mm<<<g1(K0C * NHID), TB, 0, stream>>>(tmpA, enc0_W, enc0_b, lat1, K0C, NHID, NHID, 2);

  // pooling level 1
  k_small_mm<<<g1(K0C * K1C), 64, 0, stream>>>(lat1, lin1_W, lin1_b, logits1, K0C, NHID, K1C, 0);
  k_gax1<<<1, 32, 0, stream>>>(logits1, c1);
  k_xc1<<<1, 256, 0, stream>>>(lat1, c1, xc1);
  k_l2norm_rows<<<K1C, 256, 0, stream>>>(xc1);
  k_adj1<<<1, 32, 0, stream>>>(adj0, c1, adj1);
  k_scale_sum<<<1, 128, 0, stream>>>(adj1, K1C * K1C);
  k_small_mm<<<g1(K1C * NHID), TB, 0, stream>>>(adj1, xc1, nullptr, tmpB, K1C, K1C, NHID, 0);
  k_small_mm<<<g1(K1C * NHID), TB, 0, stream>>>(tmpB, enc1_W, enc1_b, lat2, K1C, NHID, NHID, 2);
  k_rep_lat<<<g1((long)NNODE * NHID), TB, 0, stream>>>(lat1, lat2, c0, c1, rep);

  // head: relu(rep@fc1+b) @ fc2 + b, then log_softmax over 16
  k_gemm<4><<<gg256, 256, 0, stream>>>(rep, REPW, fc1w, 256, fc1_b, nullptr, 0, mid2, 256, NNODE, 256, 768, 1);
  k_gemm<1><<<gg16, 256, 0, stream>>>(mid2, 256, fc2w, 16, fc2_b, out16, 16, nullptr, 0, NNODE, 16, 256, 0);
  k_lsm16<<<g1(NNODE), TB, 0, stream>>>(out16, out, NNODE);
}
